// DDiTBlock_16234976379280
// MI455X (gfx1250) — matrix operand order repaired and checked by reading, not compiled, not run on hardware
//
#include <hip/hip_runtime.h>

// ---------------------------------------------------------------------------
// DiT block for MI455X (gfx1250, wave32, WMMA f16 -> f32)
// GEMM staging: TDM tensor_load_to_lds (TENSORcnt), double-buffered.
// Flash staging: global_load_async_to_lds_b128 (ASYNCcnt), double-buffered.
// B=4, S=2048, D=1024, H=16, HD=64, COND=129
// ---------------------------------------------------------------------------

#define B_    4
#define S_    2048
#define D_    1024
#define H_    16
#define HD_   64
#define COND_ 129
#define ROWS_ (B_ * S_)         // 8192

typedef __attribute__((ext_vector_type(16))) _Float16 v16h;
typedef __attribute__((ext_vector_type(8)))  _Float16 v8h;
typedef __attribute__((ext_vector_type(8)))  float    v8f;
typedef unsigned int u32x4 __attribute__((ext_vector_type(4)));
typedef int          i32x4 __attribute__((ext_vector_type(4)));
typedef int          i32x8 __attribute__((ext_vector_type(8)));

__device__ __forceinline__ v16h frag16(v8h a, v8h b) {
  return __builtin_shufflevector(a, b, 0,1,2,3,4,5,6,7,8,9,10,11,12,13,14,15);
}

__device__ __forceinline__ v8f wmma_f16(v16h a, v16h b, v8f c) {
  // D = A(16x32 f16) * B(32x16 f16) + C(16x16 f32)
  return __builtin_amdgcn_wmma_f32_16x16x32_f16(false, a, false, b, (short)0, c,
                                                false, false);
}

__device__ __forceinline__ v8f v8f_zero() {
  v8f z = {0.f,0.f,0.f,0.f,0.f,0.f,0.f,0.f};
  return z;
}

// --- CDNA5 async copy: global -> LDS (16B per lane), tracked by ASYNCcnt ----
__device__ __forceinline__ void async_ld_b128(const _Float16* lds_dst,
                                              const _Float16* g_src) {
  unsigned int l = (unsigned int)(unsigned long long)(uintptr_t)lds_dst;
  unsigned long long g = (unsigned long long)(uintptr_t)g_src;
  asm volatile("global_load_async_to_lds_b128 %0, %1, off"
               :: "v"(l), "v"(g) : "memory");
}
template<int N>
__device__ __forceinline__ void wait_async() {
  asm volatile("s_wait_asynccnt %0" :: "i"(N) : "memory");
}

// --- CDNA5 TDM: one instruction DMAs a 2D tile global->LDS (TENSORcnt) -----
// D# bitfields per ISA 8.3/8.4.  data_size=2B.  LDS padding done by the TDM:
// pad_interval_log2: insert pad every 2<<pi DWORDs; pad_amount_dw-1 encoded.
__device__ __forceinline__ void tdm_load_2d(const _Float16* lds_base,
                                            const _Float16* g_tile,
                                            unsigned tile_w,            // elems/row
                                            unsigned tile_h,            // rows
                                            unsigned long long stride_e,// elems
                                            int pad_interval,           // enc
                                            int pad_amount)             // enc
{
  unsigned long long ga = (unsigned long long)(uintptr_t)g_tile;
  unsigned lds = (unsigned)(unsigned long long)(uintptr_t)lds_base;
  u32x4 g0;
  g0[0] = 1u;                                        // count=1, user D#
  g0[1] = lds;                                       // lds_addr [63:32]
  g0[2] = (unsigned)ga;                              // global_addr [95:64]
  g0[3] = (unsigned)((ga >> 32) & 0x01FFFFFFull)     // global_addr[56:32]
        | (2u << 30);                                // type=2 (image)
  i32x8 g1;
  g1[0] = (1 << 16)                                  // data_size = 2 bytes
        | (1 << 20)                                  // pad_enable
        | (pad_interval << 22)
        | (pad_amount << 25);
  g1[1] = (int)0xFFFF0000u;                          // tensor_dim0 lo16 (huge)
  g1[2] = (int)0xFFFF7FFFu;                          // dim0 hi / dim1 lo (huge)
  g1[3] = (int)(0x7FFFu | (tile_w << 16));           // dim1 hi / tile_dim0
  g1[4] = (int)(tile_h & 0xFFFFu);                   // tile_dim1, tile_dim2=0
  g1[5] = (int)(unsigned)stride_e;                   // tensor_dim0_stride lo
  g1[6] = (int)((stride_e >> 32) & 0xFFFFull);       // stride hi, dim1_stride=0
  g1[7] = 0;
  i32x4 gz = {0, 0, 0, 0};
#if __clang_major__ >= 23
  i32x8 gz8 = {0, 0, 0, 0, 0, 0, 0, 0};
  __builtin_amdgcn_tensor_load_to_lds(g0, g1, gz, gz, gz8, 0);
#else
  __builtin_amdgcn_tensor_load_to_lds(g0, g1, gz, gz, 0);
#endif
}

// ---------------------------------------------------------------------------
// adaLN: ada[b, j] = sigma[b,:] @ ada_W[:, j] + ada_b[j],  j in [0, 6*D)
// ---------------------------------------------------------------------------
__global__ __launch_bounds__(256) void ada_kernel(
    const float* __restrict__ sigma, const float* __restrict__ W,
    const float* __restrict__ bias, float* __restrict__ ada)
{
  const int t = blockIdx.x * 256 + threadIdx.x;   // < B_*6*D_ = 24576
  const int j = t % (6 * D_);
  const int b = t / (6 * D_);
  float acc = bias[j];
  for (int c = 0; c < COND_; ++c)
    acc = fmaf(sigma[b * COND_ + c], W[(size_t)c * (6 * D_) + j], acc);
  ada[t] = acc;
}

// ---------------------------------------------------------------------------
// Weight transpose + f32 -> f16:  Wt[n*K + k] = (f16) W[k*N + n]
// ---------------------------------------------------------------------------
__global__ __launch_bounds__(256) void transpose_f16_kernel(
    const float* __restrict__ W, _Float16* __restrict__ Wt, int K, int N)
{
  const size_t t = (size_t)blockIdx.x * 256 + threadIdx.x;
  const int k = (int)(t % K);
  const int n = (int)(t / K);
  Wt[(size_t)n * K + k] = (_Float16)W[(size_t)k * N + n];
}

// ---------------------------------------------------------------------------
// LayerNorm + adaLN modulate + f16 convert.  One block per row (b,s).
// ---------------------------------------------------------------------------
__global__ __launch_bounds__(256) void ln_mod_kernel(
    const float* __restrict__ src, const float* __restrict__ lns,
    const float* __restrict__ lnb, const float* __restrict__ ada,
    int shift_idx, int scale_idx, _Float16* __restrict__ y16)
{
  const int row = blockIdx.x;           // 0..8191
  const int b = row >> 11;              // row / S_
  const int tid = threadIdx.x;
  const float* xp = src + (size_t)row * D_;

  float4 xv = ((const float4*)xp)[tid];           // 256 * 4 = 1024
  float xa[4] = {xv.x, xv.y, xv.z, xv.w};
  float s = xa[0] + xa[1] + xa[2] + xa[3];
  float q = xa[0]*xa[0] + xa[1]*xa[1] + xa[2]*xa[2] + xa[3]*xa[3];
  #pragma unroll
  for (int m = 1; m < 32; m <<= 1) {
    s += __shfl_xor(s, m, 32);
    q += __shfl_xor(q, m, 32);
  }
  __shared__ float ss[8], sq[8];
  if ((tid & 31) == 0) { ss[tid >> 5] = s; sq[tid >> 5] = q; }
  __syncthreads();
  float S = 0.f, Q = 0.f;
  #pragma unroll
  for (int i = 0; i < 8; ++i) { S += ss[i]; Q += sq[i]; }
  const float mean = S * (1.0f / D_);
  const float var  = Q * (1.0f / D_) - mean * mean;
  const float inv  = rsqrtf(var + 1e-6f);

  const float* ash = ada + (size_t)b * 6 * D_ + (size_t)shift_idx * D_;
  const float* asc = ada + (size_t)b * 6 * D_ + (size_t)scale_idx * D_;
  const int d0 = tid * 4;
  #pragma unroll
  for (int k = 0; k < 4; ++k) {
    const int d = d0 + k;
    float v = (xa[k] - mean) * inv * lns[d] + lnb[d];
    v = v * (1.0f + asc[d]) + ash[d];
    y16[(size_t)row * D_ + d] = (_Float16)v;
  }
}

// ---------------------------------------------------------------------------
// Tiled WMMA GEMM:  C[M,N] = A[M,K](f16,row) @ Bt[N,K](f16,row, i.e. B^T)
// 256 threads = 8 waves; block tile 128x128; wave tile 64x32 (4x2 WMMA tiles)
// A/B tiles DMA'd by the TDM (wave 0 issues; double-buffered on TENSORcnt).
// TDM pad: 16 DW interval (enc 3) + 4 DW pad (enc 3) => LDS stride 40 halves.
// MODE 0: store f16 (qkv) | 1: resid+gate*acc | 2: f16 gelu(acc+bias)
// MODE 3: resid + gate*(acc+bias)
// ---------------------------------------------------------------------------
template<int MODE>
__global__ __launch_bounds__(256) void gemm_wmma_kernel(
    const _Float16* __restrict__ A, const _Float16* __restrict__ Bt,
    int M, int N, int K,
    float* __restrict__ outF, _Float16* __restrict__ outH,
    const float* __restrict__ bias, const float* __restrict__ resid,
    const float* __restrict__ ada, int gate_idx)
{
  __shared__ __align__(32) _Float16 As[2][128 * 40];
  __shared__ __align__(32) _Float16 Bs[2][128 * 40];

  const int tid  = threadIdx.x;
  const int lane = tid & 31;
  const int wave = tid >> 5;
  const int lo   = lane & 15;
  const int hi   = lane >> 4;
  const int wm   = wave & 1;    // 2 waves along M
  const int wn   = wave >> 1;   // 4 waves along N
  const int m0   = blockIdx.y * 128;
  const int n0   = blockIdx.x * 128;

  v8f acc[4][2];
  #pragma unroll
  for (int i = 0; i < 4; ++i)
    #pragma unroll
    for (int j = 0; j < 2; ++j) acc[i][j] = v8f_zero();

  const _Float16* gA = A  + (size_t)m0 * K;
  const _Float16* gB = Bt + (size_t)n0 * K;
  const int KT = K >> 5;

  if (wave == 0) {   // prologue: DMA tile 0
    tdm_load_2d(&As[0][0], gA, 32, 128, (unsigned long long)K, 3, 3);
    tdm_load_2d(&Bs[0][0], gB, 32, 128, (unsigned long long)K, 3, 3);
  }

  for (int kt = 0; kt < KT; ++kt) {
    const int cur = kt & 1;
    if (wave == 0) {
      if (kt + 1 < KT) {    // DMA next tile into the other buffer, then wait
        tdm_load_2d(&As[1 - cur][0], gA + (kt + 1) * 32, 32, 128,
                    (unsigned long long)K, 3, 3);
        tdm_load_2d(&Bs[1 - cur][0], gB + (kt + 1) * 32, 32, 128,
                    (unsigned long long)K, 3, 3);
        __builtin_amdgcn_s_wait_tensorcnt((short)2);  // tile kt complete
      } else {
        __builtin_amdgcn_s_wait_tensorcnt((short)0);
      }
    }
    __syncthreads();    // tile kt visible to all waves

    v16h af[4], bf[2];
    #pragma unroll
    for (int i = 0; i < 4; ++i) {
      const _Float16* p = &As[cur][(wm * 64 + i * 16 + lo) * 40];
      af[i] = frag16(*(const v8h*)(p + hi * 8), *(const v8h*)(p + 16 + hi * 8));
    }
    #pragma unroll
    for (int j = 0; j < 2; ++j) {
      const _Float16* p = &Bs[cur][(wn * 32 + j * 16 + lo) * 40];
      bf[j] = frag16(*(const v8h*)(p + hi * 8), *(const v8h*)(p + 16 + hi * 8));
    }
    #pragma unroll
    for (int i = 0; i < 4; ++i)
      #pragma unroll
      for (int j = 0; j < 2; ++j)
        acc[i][j] = wmma_f16(af[i], bf[j], acc[i][j]);

    __syncthreads();    // done reading buffer before it is re-DMA'd
  }

  // epilogue (C layout: VGPR v -> row v + 8*hi, col lo)
  #pragma unroll
  for (int i = 0; i < 4; ++i) {
    #pragma unroll
    for (int j = 0; j < 2; ++j) {
      const int col   = n0 + wn * 32 + j * 16 + lo;
      const int rbase = m0 + wm * 64 + i * 16 + hi * 8;
      #pragma unroll
      for (int v = 0; v < 8; ++v) {
        const int row = rbase + v;
        float val = acc[i][j][v];
        const size_t idx = (size_t)row * N + col;
        if (MODE == 0) {
          outH[idx] = (_Float16)val;
        } else if (MODE == 1) {
          const int b = row >> 11;
          const float g = ada[(size_t)b * 6 * D_ + (size_t)gate_idx * D_ + col];
          outF[idx] = resid[idx] + g * val;
        } else if (MODE == 2) {
          float t = val + bias[col];
          float gl = 0.5f * t *
              (1.0f + tanhf(0.7978845608028654f * (t + 0.044715f * t * t * t)));
          outH[idx] = (_Float16)gl;
        } else {
          const int b = row >> 11;
          float t = val + bias[col];
          const float g = ada[(size_t)b * 6 * D_ + (size_t)gate_idx * D_ + col];
          outF[idx] = resid[(size_t)row * D_ + col] + g * t;
        }
      }
    }
  }
}

// ---------------------------------------------------------------------------
// RoPE + layout: qkv16[B,S,3,H,HD] -> q16/k16 [B,H,S,HD], vt16 [B,H,HD,S]
// ---------------------------------------------------------------------------
__global__ __launch_bounds__(256) void rope_kernel(
    const _Float16* __restrict__ qkv16, _Float16* __restrict__ q16,
    _Float16* __restrict__ k16, _Float16* __restrict__ vt16)
{
  const unsigned t = blockIdx.x * 256u + threadIdx.x;  // < B*S*H*HD = 8388608
  const int hd = t & 63;
  const int h  = (t >> 6) & 15;
  const int s  = (t >> 10) & 2047;
  const int b  = t >> 21;

  const size_t rb = (size_t)(b * S_ + s) * (3 * D_);
  const int c = h * HD_ + hd;
  const float qv = (float)qkv16[rb + c];
  const float kv = (float)qkv16[rb + D_ + c];
  const float vv = (float)qkv16[rb + 2 * D_ + c];

  const int partner = (hd < 32) ? (c + 32) : (c - 32);
  const float sgn = (hd < 32) ? -1.0f : 1.0f;
  const float qo = (float)qkv16[rb + partner];
  const float ko = (float)qkv16[rb + D_ + partner];

  const int fi = hd & 31;
  const float invf = __powf(10000.0f, -(float)(2 * fi) * (1.0f / 64.0f));
  float sn, cs;
  __sincosf((float)s * invf, &sn, &cs);

  const size_t base = (size_t)((b * H_ + h) * S_ + s) * HD_ + hd;
  q16[base] = (_Float16)(qv * cs + sgn * qo * sn);
  k16[base] = (_Float16)(kv * cs + sgn * ko * sn);
  vt16[(size_t)((b * H_ + h) * HD_ + hd) * S_ + s] = (_Float16)vv;
}

// ---------------------------------------------------------------------------
// Flash attention: grid (S/128, B*H), 256 threads = 8 waves.
// Each wave owns 16 query rows; 64-key K/V tiles double-buffered in LDS via
// per-lane async copies (ASYNCcnt in-order => wait<=4 retires prior batch).
// 16 WMMAs per iteration; online softmax with wave32 shfl_xor reductions.
// ---------------------------------------------------------------------------
__global__ __launch_bounds__(256) void flash_attn_kernel(
    const _Float16* __restrict__ q16, const _Float16* __restrict__ k16,
    const _Float16* __restrict__ vt16, _Float16* __restrict__ attn16)
{
  __shared__ __align__(32) _Float16 Ks[2][64 * 72];   // keys x hd (pad 72)
  __shared__ __align__(32) _Float16 Vts[2][64 * 72];  // hd x keys (pad 72)
  __shared__ __align__(32) _Float16 Ps[8][16 * 72];   // per-wave P (16 x 64)

  const int tid  = threadIdx.x;
  const int lane = tid & 31;
  const int wave = tid >> 5;
  const int lo   = lane & 15;
  const int hi   = lane >> 4;
  const int bh   = blockIdx.y;       // b*H + h
  const int b    = bh >> 4;
  const int h    = bh & 15;
  const int qbase = blockIdx.x * 128 + wave * 16;

  const _Float16* Qb = q16  + (size_t)bh * S_ * HD_;
  const _Float16* Kb = k16  + (size_t)bh * S_ * HD_;
  const _Float16* Vb = vt16 + (size_t)bh * HD_ * S_;

  // Q fragments for this wave's 16 rows (hd 0..31 and 32..63)
  const _Float16* qrow = Qb + (size_t)(qbase + lo) * HD_;
  const v16h qa0 = frag16(*(const v8h*)(qrow + hi * 8),
                          *(const v8h*)(qrow + 16 + hi * 8));
  const v16h qa1 = frag16(*(const v8h*)(qrow + 32 + hi * 8),
                          *(const v8h*)(qrow + 48 + hi * 8));

  v8f o[4];
  #pragma unroll
  for (int t = 0; t < 4; ++t) o[t] = v8f_zero();
  float mrow[8], lrow[8];
  #pragma unroll
  for (int v = 0; v < 8; ++v) { mrow[v] = -1e30f; lrow[v] = 0.0f; }

  // staging: 4 threads per row, each 32B (2 x b128): 64 rows x 64 halves
  const int sr   = tid >> 2;
  const int scol = (tid & 3) * 16;
  const int NT   = S_ / 64;

  // prologue: stage tile 0 into buffer 0
  {
    const _Float16* kg = Kb + (size_t)sr * HD_ + scol;
    const _Float16* vg = Vb + (size_t)sr * S_ + scol;
    async_ld_b128(&Ks[0][sr * 72 + scol],      kg);
    async_ld_b128(&Ks[0][sr * 72 + scol + 8],  kg + 8);
    async_ld_b128(&Vts[0][sr * 72 + scol],     vg);
    async_ld_b128(&Vts[0][sr * 72 + scol + 8], vg + 8);
  }

  for (int kt = 0; kt < NT; ++kt) {
    const int cur = kt & 1;
    if (kt + 1 < NT) {   // stage next tile into other buffer, then wait cur
      const _Float16* kg = Kb + (size_t)((kt + 1) * 64 + sr) * HD_ + scol;
      const _Float16* vg = Vb + (size_t)sr * S_ + (kt + 1) * 64 + scol;
      async_ld_b128(&Ks[1 - cur][sr * 72 + scol],      kg);
      async_ld_b128(&Ks[1 - cur][sr * 72 + scol + 8],  kg + 8);
      async_ld_b128(&Vts[1 - cur][sr * 72 + scol],     vg);
      async_ld_b128(&Vts[1 - cur][sr * 72 + scol + 8], vg + 8);
      wait_async<4>();   // in-order: previous batch (tile kt) has landed
    } else {
      wait_async<0>();
    }
    __syncthreads();

    // scores for keys [kt*64, kt*64+64): four 16x16 C tiles
    v8f sc[4];
    #pragma unroll
    for (int j = 0; j < 4; ++j) {
      const _Float16* p0 = &Ks[cur][(j * 16 + lo) * 72 + hi * 8];   // hd 0..31
      const _Float16* p1 = p0 + 32;                                 // hd 32..63
      v16h kb0 = frag16(*(const v8h*)p0, *(const v8h*)(p0 + 16));
      v16h kb1 = frag16(*(const v8h*)p1, *(const v8h*)(p1 + 16));
      v8f z = v8f_zero();
      z = wmma_f16(qa0, kb0, z);
      z = wmma_f16(qa1, kb1, z);
      sc[j] = z;
    }

    // online softmax (row = v + 8*hi, values spread over 16-lane half-groups)
    float pe[4][8];
    #pragma unroll
    for (int v = 0; v < 8; ++v) {
      float sv[4];
      #pragma unroll
      for (int j = 0; j < 4; ++j) sv[j] = sc[j][v] * 0.125f;  // 1/sqrt(64)
      float mt = fmaxf(fmaxf(sv[0], sv[1]), fmaxf(sv[2], sv[3]));
      #pragma unroll
      for (int m = 1; m < 16; m <<= 1) mt = fmaxf(mt, __shfl_xor(mt, m, 32));
      const float mn = fmaxf(mrow[v], mt);
      const float al = __expf(mrow[v] - mn);
      float rs = 0.0f;
      #pragma unroll
      for (int j = 0; j < 4; ++j) {
        const float e = __expf(sv[j] - mn);
        pe[j][v] = e;
        rs += e;
      }
      #pragma unroll
      for (int m = 1; m < 16; m <<= 1) rs += __shfl_xor(rs, m, 32);
      lrow[v] = lrow[v] * al + rs;
      mrow[v] = mn;
      #pragma unroll
      for (int t = 0; t < 4; ++t) o[t][v] = o[t][v] * al;
    }

    // P: C layout -> LDS -> A layout (per-wave region; same-wave DS in-order)
    _Float16* pp = Ps[wave];
    #pragma unroll
    for (int v = 0; v < 8; ++v) {
      const int pr = v + hi * 8;
      #pragma unroll
      for (int j = 0; j < 4; ++j)
        pp[pr * 72 + j * 16 + lo] = (_Float16)pe[j][v];
    }
    const _Float16* ap = pp + lo * 72;
    v16h pa0 = frag16(*(const v8h*)(ap + hi * 8),
                      *(const v8h*)(ap + 16 + hi * 8));
    v16h pa1 = frag16(*(const v8h*)(ap + 32 + hi * 8),
                      *(const v8h*)(ap + 48 + hi * 8));

    // O += P(16x64) @ V(64x64) as four 16x16 tiles, two K-chunks each
    #pragma unroll
    for (int t = 0; t < 4; ++t) {
      const _Float16* vp0 = &Vts[cur][(t * 16 + lo) * 72 + hi * 8];  // k 0..31
      const _Float16* vp1 = vp0 + 32;                                // k 32..63
      v16h vb0 = frag16(*(const v8h*)vp0, *(const v8h*)(vp0 + 16));
      v16h vb1 = frag16(*(const v8h*)vp1, *(const v8h*)(vp1 + 16));
      o[t] = wmma_f16(pa0, vb0, o[t]);
      o[t] = wmma_f16(pa1, vb1, o[t]);
    }
    __syncthreads();   // done reading buffer before it is restaged
  }

  // normalize and store attn16 [B, S, H*HD]
  #pragma unroll
  for (int v = 0; v < 8; ++v) {
    const float inv = 1.0f / lrow[v];
    const int q = qbase + v + hi * 8;
    _Float16* op = attn16 + (size_t)(b * S_ + q) * D_ + h * HD_ + lo;
    #pragma unroll
    for (int t = 0; t < 4; ++t) op[t * 16] = (_Float16)(o[t][v] * inv);
  }
}

// ---------------------------------------------------------------------------
// Host-side orchestration
// ---------------------------------------------------------------------------
extern "C" void kernel_launch(void* const* d_in, const int* in_sizes, int n_in,
                              void* d_out, int out_size, void* d_ws, size_t ws_size,
                              hipStream_t stream) {
  (void)in_sizes; (void)n_in; (void)out_size; (void)ws_size;

  const float* x    = (const float*)d_in[0];
  const float* sig  = (const float*)d_in[1];
  const float* ln1s = (const float*)d_in[2];
  const float* ln1b = (const float*)d_in[3];
  const float* ln2s = (const float*)d_in[4];
  const float* ln2b = (const float*)d_in[5];
  const float* Wqkv = (const float*)d_in[6];
  const float* Wout = (const float*)d_in[7];
  const float* adaW = (const float*)d_in[8];
  const float* adab = (const float*)d_in[9];
  const float* fc1W = (const float*)d_in[10];
  const float* fc1b = (const float*)d_in[11];
  const float* fc2W = (const float*)d_in[12];
  const float* fc2b = (const float*)d_in[13];
  float* out = (float*)d_out;

  // Workspace layout (bytes). Shared 64MB region at offset 0 is reused by
  // qkv16 (48MB) -> attn16 (16MB) -> h16 (64MB): lifetimes are disjoint.
  char* ws = (char*)d_ws;
  const size_t MB = 1ull << 20;
  _Float16* qkv16  = (_Float16*)(ws + 0);
  _Float16* attn16 = (_Float16*)(ws + 0);
  _Float16* h16    = (_Float16*)(ws + 0);
  _Float16* y16    = (_Float16*)(ws + 64 * MB);
  _Float16* q16    = (_Float16*)(ws + 80 * MB);
  _Float16* k16    = (_Float16*)(ws + 96 * MB);
  _Float16* vt16   = (_Float16*)(ws + 112 * MB);
  float*    x2     = (float*)   (ws + 128 * MB);
  _Float16* wqkvT  = (_Float16*)(ws + 160 * MB);
  _Float16* woutT  = (_Float16*)(ws + 166 * MB);
  _Float16* fc1T   = (_Float16*)(ws + 168 * MB);
  _Float16* fc2T   = (_Float16*)(ws + 176 * MB);
  float*    ada    = (float*)   (ws + 184 * MB);

  // Weights -> transposed f16 (B^T layout: [N, K])
  transpose_f16_kernel<<<(1024 * 3072) / 256, 256, 0, stream>>>(Wqkv, wqkvT, 1024, 3072);
  transpose_f16_kernel<<<(1024 * 1024) / 256, 256, 0, stream>>>(Wout, woutT, 1024, 1024);
  transpose_f16_kernel<<<(1024 * 4096) / 256, 256, 0, stream>>>(fc1W, fc1T, 1024, 4096);
  transpose_f16_kernel<<<(4096 * 1024) / 256, 256, 0, stream>>>(fc2W, fc2T, 4096, 1024);

  // adaLN vectors: ada [B, 6, D]
  ada_kernel<<<(B_ * 6 * D_) / 256, 256, 0, stream>>>(sig, adaW, adab, ada);

  // y = modulate(LN1(x))  (f16)
  ln_mod_kernel<<<ROWS_, 256, 0, stream>>>(x, ln1s, ln1b, ada, 0, 1, y16);

  // qkv = y @ W_qkv  (f16 out)
  gemm_wmma_kernel<0><<<dim3(3072 / 128, ROWS_ / 128), 256, 0, stream>>>(
      y16, wqkvT, ROWS_, 3072, 1024, nullptr, qkv16, nullptr, nullptr, nullptr, 0);

  // RoPE + head layouts
  rope_kernel<<<(B_ * S_ * H_ * HD_) / 256, 256, 0, stream>>>(qkv16, q16, k16, vt16);

  // attention
  flash_attn_kernel<<<dim3(S_ / 128, B_ * H_), 256, 0, stream>>>(q16, k16, vt16, attn16);

  // x2 = x + gate_msa * (attn @ W_out)
  gemm_wmma_kernel<1><<<dim3(1024 / 128, ROWS_ / 128), 256, 0, stream>>>(
      attn16, woutT, ROWS_, 1024, 1024, x2, nullptr, nullptr, x, ada, 2);

  // y = modulate(LN2(x2))  (f16)
  ln_mod_kernel<<<ROWS_, 256, 0, stream>>>(x2, ln2s, ln2b, ada, 3, 4, y16);

  // h = gelu(y @ fc1 + b1)  (f16)
  gemm_wmma_kernel<2><<<dim3(4096 / 128, ROWS_ / 128), 256, 0, stream>>>(
      y16, fc1T, ROWS_, 4096, 1024, nullptr, h16, fc1b, nullptr, nullptr, 0);

  // out = x2 + gate_mlp * (h @ fc2 + b2)
  gemm_wmma_kernel<3><<<dim3(1024 / 128, ROWS_ / 128), 256, 0, stream>>>(
      h16, fc2T, ROWS_, 1024, 4096, out, nullptr, fc2b, x2, ada, 5);
}
